// RefinementStep_25065429139871
// MI455X (gfx1250) — matrix operand order repaired and checked by reading, not compiled, not run on hardware
//
#include <hip/hip_runtime.h>
#include <hip/hip_bf16.h>
#include <stdint.h>

// ---------------- types ----------------
typedef __attribute__((ext_vector_type(16))) __bf16          v16bf;
typedef __attribute__((ext_vector_type(8)))  float           v8f;
typedef __attribute__((ext_vector_type(16))) unsigned short  u16x16;

#define NI 152      // B*NJ images through the crop CNN
#define BJ 38
#define NB 4

__device__ __forceinline__ unsigned short f2bf(float f) {
  unsigned u = __float_as_uint(f);
  unsigned r = u + 0x7FFFu + ((u >> 16) & 1u);   // round-to-nearest-even
  return (unsigned short)(r >> 16);
}

// ---------------- weight prep: f32 [Cout][Cin][3][3] -> bf16 WMMA-B layout ----------------
// layout: element e = (kb*Cout + n)*32 + q,  k = kb*32 + q  (q = half*16 + j)
__global__ void k_prep(const float* __restrict__ w, unsigned short* __restrict__ wB,
                       int Cin, int Cout) {
  int K  = Cin * 9;
  int KB = (K + 31) / 32;
  int total = KB * 32 * Cout;
  for (int e = blockIdx.x * blockDim.x + threadIdx.x; e < total;
       e += gridDim.x * blockDim.x) {
    int kb  = e / (Cout * 32);
    int rem = e - kb * Cout * 32;
    int n   = rem >> 5;
    int q   = rem & 31;
    int k   = kb * 32 + q;
    float v = 0.f;
    if (k < K) v = w[(n * Cin + (k / 9)) * 9 + (k % 9)];
    wB[e] = f2bf(v);
  }
}

__global__ void k_zero(float* __restrict__ p, int n) {
  for (int i = blockIdx.x * blockDim.x + threadIdx.x; i < n;
       i += gridDim.x * blockDim.x) p[i] = 0.f;
}

// ---------------- bilinear 64->32 resize of crops, store bf16 [img][3][1024] ----------------
__global__ void k_resize(const float* __restrict__ crop, unsigned short* __restrict__ act) {
  int idx = blockIdx.x * blockDim.x + threadIdx.x;
  int total = NI * 3 * 1024;
  if (idx >= total) return;
  int img = idx / (3 * 1024);
  int rem = idx - img * 3 * 1024;
  int ch  = rem >> 10;
  int pix = rem & 1023;
  int o = pix >> 5, pc = pix & 31;
  float sy = o  * (63.f / 31.f); int y0 = (int)floorf(sy); if (y0 > 62) y0 = 62; float fy = sy - y0;
  float sx = pc * (63.f / 31.f); int x0 = (int)floorf(sx); if (x0 > 62) x0 = 62; float fx = sx - x0;
  const float* base = crop + (size_t)img * 4096 * 3;   // [h][w][ch]
  #define PX(h, w) base[((h) * 64 + (w)) * 3 + ch]
  float v = (1.f - fy) * ((1.f - fx) * PX(y0, x0)     + fx * PX(y0, x0 + 1)) +
            fy        * ((1.f - fx) * PX(y0 + 1, x0) + fx * PX(y0 + 1, x0 + 1));
  #undef PX
  act[((size_t)img * 3 + ch) * 1024 + pix] = f2bf(v);
}

// ---------------- implicit-GEMM conv3x3 SAME on 32x32, bf16 WMMA, f32 accum ----------------
template <int CIN, int COUT, bool POOL>
__global__ void __launch_bounds__(128) k_conv(
    const unsigned short* __restrict__ actIn,   // bf16 [img][CIN][1024]
    const unsigned short* __restrict__ wB,      // bf16 B-fragment layout
    const float* __restrict__ cb, const float* __restrict__ bgm,
    const float* __restrict__ bbt,
    unsigned short* __restrict__ actOut,        // bf16 [img][COUT][1024]
    float* __restrict__ pooled)                 // f32  [img][COUT] (POOL only)
{
  constexpr int K   = CIN * 9;
  constexpr int KB  = (K + 31) / 32;
  constexpr int NT  = COUT / 16;
  constexpr int TPW = NT / 4;

  __shared__ unsigned short At[16][32];
  const int tid  = threadIdx.x;
  const int wave = tid >> 5, lane = tid & 31;
  const int lrow = lane & 15, lhi = lane >> 4;
  const int m0   = blockIdx.x * 16;
  const int img  = blockIdx.y;

  v8f acc[TPW] = {};

  for (int kb = 0; kb < KB; ++kb) {
    __syncthreads();
    for (int e = tid; e < 512; e += 128) {        // stage A tile (im2col) into LDS
      int m = e >> 5, kk = e & 31;
      int k = kb * 32 + kk;
      unsigned short v = 0;
      if (k < K) {
        int cin = k / 9, r = k - cin * 9;
        int ky = r / 3, kx = r - ky * 3;
        int mrow = m0 + m;
        int y = (mrow >> 5) + ky - 1;
        int x = (mrow & 31) + kx - 1;
        if ((unsigned)y < 32u && (unsigned)x < 32u)
          v = actIn[((size_t)img * CIN + cin) * 1024 + y * 32 + x];
      }
      At[m][kk] = v;
    }
    __syncthreads();

    // A fragment: lane l%16 = row, K halves split by l/16 (contiguous k = 16*(l/16)+s, same order as wB)
    union { u16x16 v; unsigned u[8]; } au;
#pragma unroll
    for (int j = 0; j < 8; ++j) {
      int kk0 = lhi * 16 + 2 * j;
      au.u[j] = *(const unsigned*)&At[lrow][kk0];
    }
    v16bf a = __builtin_bit_cast(v16bf, au.v);

    if (kb + 1 < KB)  // global_prefetch_b8 of next K-block's weights
      __builtin_prefetch(wB + ((size_t)(kb + 1) * COUT + wave * TPW * 16 + lrow) * 32, 0, 3);

#pragma unroll
    for (int t = 0; t < TPW; ++t) {
      int n = (wave * TPW + t) * 16 + lrow;
      const u16x16* bp =
          (const u16x16*)(wB + ((size_t)(kb * COUT + n)) * 32 + lhi * 16);
      v16bf bf = __builtin_bit_cast(v16bf, *bp);
      acc[t] = __builtin_amdgcn_wmma_f32_16x16x32_bf16(
          false, a, false, bf, (short)0, acc[t], false, false);
    }
  }

  const float inv = 0.9999950000374997f;  // 1/sqrt(1+1e-5)
#pragma unroll
  for (int t = 0; t < TPW; ++t) {
    int n = (wave * TPW + t) * 16 + lrow;
    float bi = cb[n], gm = bgm[n] * inv, bt = bbt[n];
    if (POOL) {
      float s = 0.f;
#pragma unroll
      for (int r = 0; r < 8; ++r) {
        float v = (acc[t][r] + bi) * gm + bt;
        s += v > 0.f ? v : 0.f;
      }
      atomicAdd(&pooled[(size_t)img * COUT + n], s * (1.0f / 1024.0f));
    } else {
#pragma unroll
      for (int r = 0; r < 8; ++r) {
        float v = (acc[t][r] + bi) * gm + bt;
        v = v > 0.f ? v : 0.f;
        int m = m0 + r + lhi * 8;
        actOut[((size_t)img * COUT + n) * 1024 + m] = f2bf(v);
      }
    }
  }
}

// ---------------- multi-scale bilinear sample + per-scale projection ----------------
__global__ void k_sample_ms(const float* __restrict__ coords,
                            const float* __restrict__ f0, const float* __restrict__ f1,
                            const float* __restrict__ f2, const float* __restrict__ f3,
                            const float* __restrict__ spw, const float* __restrict__ spb,
                            float* __restrict__ msf) {
  __shared__ float sm[4][256];
  int j = blockIdx.x, b = blockIdx.y, c = threadIdx.x;
  float cx = coords[(b * BJ + j) * 2 + 0];
  float cy = coords[(b * BJ + j) * 2 + 1];
  const float* fs[4] = { f0, f1, f2, f3 };
  const int SZ[4] = { 64, 32, 16, 8 };
  for (int k = 0; k < 4; ++k) {
    int W = SZ[k];
    float x = cx * (1.0f / 64.0f) * (W - 1);
    float y = cy * (1.0f / 64.0f) * (W - 1);
    int x0 = (int)floorf(x), y0 = (int)floorf(y);
    const float* fb = fs[k] + ((size_t)b * 256 + c) * W * W;
    float accv = 0.f;
    for (int dy = 0; dy < 2; ++dy)
      for (int dx = 0; dx < 2; ++dx) {
        int xi = x0 + dx, yi = y0 + dy;
        float w = (1.f - fabsf(x - (float)xi)) * (1.f - fabsf(y - (float)yi));
        if (!(xi >= 0 && xi <= W - 1 && yi >= 0 && yi <= W - 1)) w = 0.f;
        int xc = xi < 0 ? 0 : (xi > W - 1 ? W - 1 : xi);
        int yc = yi < 0 ? 0 : (yi > W - 1 ? W - 1 : yi);
        accv += w * fb[yc * W + xc];
      }
    sm[k][c] = accv;
  }
  __syncthreads();
  float out = 0.f;
  for (int k = 0; k < 4; ++k) {
    float d = spb[k * 256 + c];
    const float* wr = spw + ((size_t)k * 256 + c) * 256;
    for (int i = 0; i < 256; ++i) d += sm[k][i] * wr[i];
    out += d;
  }
  msf[((size_t)b * BJ + j) * 256 + c] = out * 0.25f;
}

// ---------------- 512->256 crop projection ----------------
__global__ void k_crop_proj(const float* __restrict__ pooled, const float* __restrict__ cw,
                            const float* __restrict__ cbias, float* __restrict__ cf) {
  int j = blockIdx.x, b = blockIdx.y, o = threadIdx.x;
  const float* pr = pooled + ((size_t)(b * BJ + j)) * 512;
  const float* wr = cw + (size_t)o * 512;
  float a = cbias[o];
  for (int i = 0; i < 512; ++i) a += pr[i] * wr[i];
  cf[((size_t)b * BJ + j) * 256 + o] = a;
}

// ---------------- fuse MLP ----------------
__global__ void k_fuse(const float* __restrict__ msf, const float* __restrict__ cf,
                       const float* __restrict__ w1, const float* __restrict__ b1,
                       const float* __restrict__ w2, const float* __restrict__ b2,
                       float* __restrict__ fused) {
  __shared__ float h[128];
  __shared__ float lg[2];
  int j = blockIdx.x, b = blockIdx.y, t = threadIdx.x;
  const float* ms = msf + ((size_t)b * BJ + j) * 256;
  const float* cr = cf  + ((size_t)b * BJ + j) * 256;
  if (t < 128) {
    float a = b1[t];
    const float* wr = w1 + (size_t)t * 512;
    for (int i = 0; i < 256; ++i) a += ms[i] * wr[i];
    for (int i = 0; i < 256; ++i) a += cr[i] * wr[256 + i];
    h[t] = a > 0.f ? a : 0.f;
  }
  __syncthreads();
  if (t < 2) {
    float a = b2[t];
    const float* wr = w2 + t * 128;
    for (int i = 0; i < 128; ++i) a += h[i] * wr[i];
    lg[t] = a;
  }
  __syncthreads();
  float m  = fmaxf(lg[0], lg[1]);
  float e0 = expf(lg[0] - m), e1 = expf(lg[1] - m);
  float w0 = e0 / (e0 + e1), wq = e1 / (e0 + e1);
  fused[((size_t)b * BJ + j) * 256 + t] = w0 * ms[t] + wq * cr[t];
}

// ---------------- one GAT layer (diagonal attention, per reference einsum) ----------------
__global__ void __launch_bounds__(256) k_gat_layer(
    const float* __restrict__ xin, int bstride, int rstride, int cin,
    const float* __restrict__ W, const float* __restrict__ A,
    const float* __restrict__ resid, float* __restrict__ xout) {
  __shared__ float xs[19][512];
  __shared__ float hs[19][256];
  __shared__ float sv[19][4], tv[19][4], ad[19][4];
  int b = blockIdx.x, t = threadIdx.x;
  for (int e = t; e < 19 * cin; e += 256) {
    int n = e / cin, c = e - n * cin;
    xs[n][c] = xin[(size_t)b * bstride + n * rstride + c];
  }
  __syncthreads();
  for (int n = 0; n < 19; ++n) {
    float a = 0.f;
    const float* wr = W + (size_t)t * cin;
    for (int k = 0; k < cin; ++k) a += xs[n][k] * wr[k];
    hs[n][t] = a;
  }
  __syncthreads();
  if (t < 76) {
    int n = t >> 2, hh = t & 3;
    float s = 0.f, tt = 0.f;
    const float* ar = A + (size_t)hh * 512;
    for (int k = 0; k < 256; ++k) {
      int base = (k >> 6) * 128 + (k & 63);
      float hv = hs[n][k];
      s  += hv * ar[base];
      tt += hv * ar[base + 64];
    }
    sv[n][hh] = s; tv[n][hh] = tt;
  }
  __syncthreads();
  if (t < 76) {
    int n = t >> 2, hh = t & 3;
    float s = sv[n][hh];
    float mx = -1e30f;
    for (int m = 0; m < 19; ++m) {
      float e = s + tv[m][hh];
      e = e >= 0.f ? e : 0.2f * e;
      mx = fmaxf(mx, e);
    }
    float den = 0.f, num = 0.f;
    for (int m = 0; m < 19; ++m) {
      float e = s + tv[m][hh];
      e = e >= 0.f ? e : 0.2f * e;
      float ex = expf(e - mx);
      den += ex;
      if (m == n) num = ex;
    }
    ad[n][hh] = num / den;
  }
  __syncthreads();
  for (int n = 0; n < 19; ++n) {
    float v = ad[n][t >> 6] * hs[n][t];
    if (resid) v += resid[((size_t)b * 19 + n) * 256 + t];
    xout[((size_t)b * 19 + n) * 256 + t] = v;
  }
}

// ---------------- cross attention + build concat inputs for cgat ----------------
__global__ void k_cross(const float* __restrict__ bg, const float* __restrict__ sg,
                        float* __restrict__ cbg, float* __restrict__ csg) {
  __shared__ float bm[19], smv[19], cr[19][19];
  int b = blockIdx.x, t = threadIdx.x;
  if (t < 19) {
    float a = 0.f, c = 0.f;
    for (int i = 0; i < 256; ++i) {
      a += bg[((size_t)b * 19 + t) * 256 + i];
      c += sg[((size_t)b * 19 + t) * 256 + i];
    }
    bm[t] = a / 256.f; smv[t] = c / 256.f;
  }
  __syncthreads();
  if (t < 19) {
    float mx = -1e30f;
    for (int s = 0; s < 19; ++s) mx = fmaxf(mx, -fabsf(bm[t] - smv[s]));
    float den = 0.f;
    for (int s = 0; s < 19; ++s) {
      float e = expf(-fabsf(bm[t] - smv[s]) - mx);
      cr[t][s] = e; den += e;
    }
    for (int s = 0; s < 19; ++s) cr[t][s] /= den;
  }
  __syncthreads();
  for (int n = 0; n < 19; ++n) {
    float bcx = 0.f;
    for (int s = 0; s < 19; ++s) bcx += cr[n][s] * sg[((size_t)b * 19 + s) * 256 + t];
    cbg[((size_t)b * 19 + n) * 512 + t]       = bg[((size_t)b * 19 + n) * 256 + t];
    cbg[((size_t)b * 19 + n) * 512 + 256 + t] = bcx;
  }
  for (int s = 0; s < 19; ++s) {
    float scx = 0.f;
    for (int n = 0; n < 19; ++n) scx += cr[n][s] * bg[((size_t)b * 19 + n) * 256 + t];
    csg[((size_t)b * 19 + s) * 512 + t]       = sg[((size_t)b * 19 + s) * 256 + t];
    csg[((size_t)b * 19 + s) * 512 + 256 + t] = scx;
  }
}

// ---------------- refine (mean over joints folds into one matvec) ----------------
__global__ void k_refine(const float* __restrict__ be, const float* __restrict__ se,
                         const float* __restrict__ rw, const float* __restrict__ rb,
                         float* __restrict__ refined) {
  __shared__ float em[256];
  int b = blockIdx.x, t = threadIdx.x;
  float s = 0.f;
  for (int n = 0; n < 19; ++n)
    s += be[((size_t)b * 19 + n) * 256 + t] + se[((size_t)b * 19 + n) * 256 + t];
  em[t] = s / 38.f;
  __syncthreads();
  if (t < 38) {
    float a = rb[t];
    for (int c = 0; c < 256; ++c) a += rw[t * 256 + c] * em[c];
    refined[b * 38 + t] = a;
  }
}

__global__ void k_broadcast(const float* __restrict__ refined, float* __restrict__ out) {
  int k = blockIdx.x, b = blockIdx.y;
  float v = refined[b * 38 + k];
  float* o = out + ((size_t)b * 38 + k) * 4096;
  for (int i = threadIdx.x; i < 4096; i += 256) o[i] = v;
}

// ---------------- host orchestration ----------------
// input order: coords, ms0..ms3, cropped0, then params (sorted keys), step_idx
enum {
  IN_COORDS = 0, IN_MS0, IN_MS1, IN_MS2, IN_MS3, IN_CROP,
  P_ADJ_ATTN_B, P_ADJ_ATTN_W, P_ADJ_PROJ_B, P_ADJ_PROJ_W,
  P_BGAT_A0, P_BGAT_A1, P_BGAT_W0, P_BGAT_W1,
  P_BN1_B, P_BN1_G, P_BN2_B, P_BN2_G, P_BN3_B, P_BN3_G, P_BN4_B, P_BN4_G,
  P_CGAT_A0, P_CGAT_A1, P_CGAT_W0, P_CGAT_W1,
  P_CONV1_B, P_CONV1_W, P_CONV2_B, P_CONV2_W, P_CONV3_B, P_CONV3_W,
  P_CONV4_B, P_CONV4_W,
  P_CROP_PROJ_B, P_CROP_PROJ_W,
  P_FW1_B, P_FW1_W, P_FW2_B, P_FW2_W,
  P_REFINE_B, P_REFINE_W,
  P_SCALE_PROJ_B, P_SCALE_PROJ_W,
  P_SGAT_A0, P_SGAT_A1, P_SGAT_W0, P_SGAT_W1,
  IN_STEP
};

extern "C" void kernel_launch(void* const* d_in, const int* in_sizes, int n_in,
                              void* d_out, int out_size, void* d_ws, size_t ws_size,
                              hipStream_t stream) {
  (void)in_sizes; (void)n_in; (void)out_size; (void)ws_size;
  const float* F = nullptr; (void)F;
  #define INF(i) ((const float*)d_in[i])

  char* ws = (char*)d_ws;
  size_t off = 0;
  auto carve = [&](size_t bytes) -> char* {
    char* p = ws + off;
    off += (bytes + 255) & ~(size_t)255;
    return p;
  };
  unsigned short* wB1  = (unsigned short*)carve((size_t)32   * 64  * 2);
  unsigned short* wB2  = (unsigned short*)carve((size_t)576  * 128 * 2);
  unsigned short* wB3  = (unsigned short*)carve((size_t)1152 * 256 * 2);
  unsigned short* wB4  = (unsigned short*)carve((size_t)2304 * 512 * 2);
  unsigned short* actA = (unsigned short*)carve((size_t)NI * 128 * 1024 * 2);
  unsigned short* actB = (unsigned short*)carve((size_t)NI * 256 * 1024 * 2);
  float* pooled  = (float*)carve((size_t)NI * 512 * 4);
  float* msf     = (float*)carve((size_t)NB * BJ * 256 * 4);
  float* cf      = (float*)carve((size_t)NB * BJ * 256 * 4);
  float* fused   = (float*)carve((size_t)NB * BJ * 256 * 4);
  float* h0      = (float*)carve((size_t)NB * 19 * 256 * 4);
  float* bgb     = (float*)carve((size_t)NB * 19 * 256 * 4);
  float* sgb     = (float*)carve((size_t)NB * 19 * 256 * 4);
  float* cbg     = (float*)carve((size_t)NB * 19 * 512 * 4);
  float* csg     = (float*)carve((size_t)NB * 19 * 512 * 4);
  float* beb     = (float*)carve((size_t)NB * 19 * 256 * 4);
  float* seb     = (float*)carve((size_t)NB * 19 * 256 * 4);
  float* refined = (float*)carve((size_t)NB * BJ * 4);

  // weight prep (bf16 fragment layout)
  k_prep<<<dim3(8),    256, 0, stream>>>(INF(P_CONV1_W), wB1, 3,   64);
  k_prep<<<dim3(288),  256, 0, stream>>>(INF(P_CONV2_W), wB2, 64,  128);
  k_prep<<<dim3(1152), 256, 0, stream>>>(INF(P_CONV3_W), wB3, 128, 256);
  k_prep<<<dim3(4608), 256, 0, stream>>>(INF(P_CONV4_W), wB4, 256, 512);
  k_zero<<<dim3((NI * 512 + 255) / 256), 256, 0, stream>>>(pooled, NI * 512);

  // crop path: resize -> conv stack (bf16 WMMA) -> pooled
  k_resize<<<dim3((NI * 3 * 1024 + 255) / 256), 256, 0, stream>>>(INF(IN_CROP), actA);
  k_conv<3,   64,  false><<<dim3(64, NI), 128, 0, stream>>>(
      actA, wB1, INF(P_CONV1_B), INF(P_BN1_G), INF(P_BN1_B), actB, nullptr);
  k_conv<64,  128, false><<<dim3(64, NI), 128, 0, stream>>>(
      actB, wB2, INF(P_CONV2_B), INF(P_BN2_G), INF(P_BN2_B), actA, nullptr);
  k_conv<128, 256, false><<<dim3(64, NI), 128, 0, stream>>>(
      actA, wB3, INF(P_CONV3_B), INF(P_BN3_G), INF(P_BN3_B), actB, nullptr);
  k_conv<256, 512, true ><<<dim3(64, NI), 128, 0, stream>>>(
      actB, wB4, INF(P_CONV4_B), INF(P_BN4_G), INF(P_BN4_B), nullptr, pooled);

  // multi-scale sampling / projections / fuse
  k_sample_ms<<<dim3(BJ, NB), 256, 0, stream>>>(
      INF(IN_COORDS), INF(IN_MS0), INF(IN_MS1), INF(IN_MS2), INF(IN_MS3),
      INF(P_SCALE_PROJ_W), INF(P_SCALE_PROJ_B), msf);
  k_crop_proj<<<dim3(BJ, NB), 256, 0, stream>>>(
      pooled, INF(P_CROP_PROJ_W), INF(P_CROP_PROJ_B), cf);
  k_fuse<<<dim3(BJ, NB), 256, 0, stream>>>(
      msf, cf, INF(P_FW1_W), INF(P_FW1_B), INF(P_FW2_W), INF(P_FW2_B), fused);

  // bone / soft GATs
  k_gat_layer<<<NB, 256, 0, stream>>>(fused,            BJ * 256, 256, 256,
      INF(P_BGAT_W0), INF(P_BGAT_A0), nullptr, h0);
  k_gat_layer<<<NB, 256, 0, stream>>>(h0,               19 * 256, 256, 256,
      INF(P_BGAT_W1), INF(P_BGAT_A1), h0, bgb);
  k_gat_layer<<<NB, 256, 0, stream>>>(fused + 19 * 256, BJ * 256, 256, 256,
      INF(P_SGAT_W0), INF(P_SGAT_A0), nullptr, h0);
  k_gat_layer<<<NB, 256, 0, stream>>>(h0,               19 * 256, 256, 256,
      INF(P_SGAT_W1), INF(P_SGAT_A1), h0, sgb);

  // cross attention + cgat
  k_cross<<<NB, 256, 0, stream>>>(bgb, sgb, cbg, csg);
  k_gat_layer<<<NB, 256, 0, stream>>>(cbg, 19 * 512, 512, 512,
      INF(P_CGAT_W0), INF(P_CGAT_A0), nullptr, h0);
  k_gat_layer<<<NB, 256, 0, stream>>>(h0,  19 * 256, 256, 256,
      INF(P_CGAT_W1), INF(P_CGAT_A1), h0, beb);
  k_gat_layer<<<NB, 256, 0, stream>>>(csg, 19 * 512, 512, 512,
      INF(P_CGAT_W0), INF(P_CGAT_A0), nullptr, h0);
  k_gat_layer<<<NB, 256, 0, stream>>>(h0,  19 * 256, 256, 256,
      INF(P_CGAT_W1), INF(P_CGAT_A1), h0, seb);

  // refine + broadcast
  k_refine<<<NB, 256, 0, stream>>>(beb, seb, INF(P_REFINE_W), INF(P_REFINE_B), refined);
  k_broadcast<<<dim3(BJ, NB), 256, 0, stream>>>(refined, (float*)d_out);
  #undef INF
}
